// Decoder_82841329206105
// MI455X (gfx1250) — compile-verified, matrix operand order checked by reading, not run on hardware
//
#include <hip/hip_runtime.h>
#include <hip/hip_bf16.h>

typedef __attribute__((ext_vector_type(16))) _Float16 v16h;
typedef __attribute__((ext_vector_type(8)))  _Float16 v8h;
typedef __attribute__((ext_vector_type(8)))  float    v8f;

#define BB    64
#define T_ENC 256
#define T_DEC 200
#define MELR  160
#define ATT   256

#define NT 1024          // decoder block: 32 wave32s
#define NW 32

#ifndef __has_builtin
#define __has_builtin(x) 0
#endif
#if defined(__AMDGCN__) && __has_builtin(__builtin_amdgcn_tensor_load_to_lds) && \
    __has_builtin(__builtin_amdgcn_s_wait_tensorcnt)
#define HAVE_TDM 1
#else
#define HAVE_TDM 0
#endif

// ---------------------------------------------------------------------------
// WMMA helpers (gfx1250, wave32).  Fragment layout per CDNA5 ISA 7.12.2:
// 16-bit A (16x32 MxK): lanes 0-15 hold K=0..7,16..23; lanes 16-31 K=8..15,24..31.
// B (32x16 KxN) mirrors with lane%16 = n.  Both are two contiguous half8 loads.
// ---------------------------------------------------------------------------
__device__ __forceinline__ v16h load_frag(const _Float16* p, int ld, int k0, int lane) {
    int r   = lane & 15;
    int off = (lane >> 4) << 3;            // 0 or 8
    const _Float16* base = p + (size_t)r * ld + k0 + off;
    v8h lo = *(const v8h*)(base);
    v8h hi = *(const v8h*)(base + 16);
    v16h o;
#pragma unroll
    for (int i = 0; i < 8; ++i) { o[i] = lo[i]; o[8 + i] = hi[i]; }
    return o;
}

__device__ __forceinline__ v8f wmma32(v16h a, v16h b, v8f c) {
    return __builtin_amdgcn_wmma_f32_16x16x32_f16(false, a, false, b, (short)0, c,
                                                  false, false);
}

__device__ __forceinline__ v8f tile_gemm(const _Float16* A, int lda,
                                         const _Float16* W, int K,
                                         int nbase, int lane) {
    v8f acc = {};
    for (int k0 = 0; k0 < K; k0 += 32) {
        v16h a = load_frag(A, lda, k0, lane);
        v16h b = load_frag(W + (size_t)nbase * K, K, k0, lane);
        acc = wmma32(a, b, acc);
    }
    return acc;
}

__device__ __forceinline__ float sigmf(float x) { return 1.f / (1.f + expf(-x)); }

// Fused GRUCell: 6 WMMA accumulators per 16-col hidden tile, nonlinearity in regs.
__device__ __forceinline__ void gru_block(const _Float16* Ax, int lda, int Kx,
                                          const _Float16* Hh,
                                          const _Float16* wih, const _Float16* whh,
                                          const float* bih, const float* bhh,
                                          float* hstate, float* dresid,
                                          int wave, int lane) {
    for (int nt = wave; nt < 16; nt += NW) {
        int cb = nt << 4;
        v8f ir = {}, iz = {}, inn = {}, hr = {}, hz = {}, hn = {};
        for (int k0 = 0; k0 < Kx; k0 += 32) {
            v16h a = load_frag(Ax, lda, k0, lane);
            ir  = wmma32(a, load_frag(wih + (size_t)cb * Kx,         Kx, k0, lane), ir);
            iz  = wmma32(a, load_frag(wih + (size_t)(256 + cb) * Kx, Kx, k0, lane), iz);
            inn = wmma32(a, load_frag(wih + (size_t)(512 + cb) * Kx, Kx, k0, lane), inn);
        }
        for (int k0 = 0; k0 < 256; k0 += 32) {
            v16h a = load_frag(Hh, 256, k0, lane);
            hr = wmma32(a, load_frag(whh + (size_t)cb * 256,         256, k0, lane), hr);
            hz = wmma32(a, load_frag(whh + (size_t)(256 + cb) * 256, 256, k0, lane), hz);
            hn = wmma32(a, load_frag(whh + (size_t)(512 + cb) * 256, 256, k0, lane), hn);
        }
        int col = cb + (lane & 15);
        int mb  = (lane >> 4) << 3;
#pragma unroll
        for (int r = 0; r < 8; ++r) {
            int   m    = mb + r;
            float rg   = sigmf(ir[r] + bih[col]       + hr[r] + bhh[col]);
            float zg   = sigmf(iz[r] + bih[256 + col] + hz[r] + bhh[256 + col]);
            float ng   = tanhf(inn[r] + bih[512 + col] + rg * (hn[r] + bhh[512 + col]));
            float hold = hstate[m * 256 + col];
            float hnew = (1.f - zg) * ng + zg * hold;
            hstate[m * 256 + col] = hnew;
            if (dresid) dresid[m * 256 + col] += hnew;
        }
    }
}

#if HAVE_TDM
// Issue a TDM 2D tile load: 16 rows x 160 f32, row stride T_DEC*MELR floats.
// D# built per CDNA5 ISA (group0/1; groups 2/3 zero).
__device__ __forceinline__ void tdm_load_frame(const float* gsrc, unsigned lds_byte_addr) {
    typedef unsigned int u32x4 __attribute__((ext_vector_type(4)));
    typedef int          i32x4 __attribute__((ext_vector_type(4)));
    typedef int          i32x8 __attribute__((ext_vector_type(8)));
    unsigned long long ga = (unsigned long long)(uintptr_t)gsrc;
    u32x4 g0;
    g0[0] = 1u;                                                   // count=1 (valid user D#)
    g0[1] = lds_byte_addr;                                        // lds_addr
    g0[2] = (unsigned)(ga & 0xFFFFFFFFu);                         // global_addr[31:0]
    g0[3] = (unsigned)((ga >> 32) & 0x01FFFFFFu) | (2u << 30);    // addr[56:32] | type=2
    i32x8 g1;
    g1[0] = (int)(2u << 16);                                      // data_size = 4B
    g1[1] = (int)((unsigned)MELR << 16);                          // tensor_dim0 = 160
    g1[2] = (int)(16u << 16);                                     // tensor_dim1 = 16
    g1[3] = (int)((unsigned)MELR << 16);                          // tile_dim0 = 160
    g1[4] = 16;                                                   // tile_dim1 = 16
    g1[5] = (int)(T_DEC * MELR);                                  // tensor_dim0_stride
    g1[6] = 0; g1[7] = 0;
    i32x4 gz = {0, 0, 0, 0};
#if __clang_major__ >= 23
    i32x8 gz8 = {0, 0, 0, 0, 0, 0, 0, 0};
    __builtin_amdgcn_tensor_load_to_lds(g0, g1, gz, gz, gz8, 0);
#else
    __builtin_amdgcn_tensor_load_to_lds(g0, g1, gz, gz, 0);
#endif
}
#endif

// ---------------------------------------------------------------------------
// Prep kernels
// ---------------------------------------------------------------------------
__global__ void __launch_bounds__(256) f32_to_f16_kernel(const float* __restrict__ s,
                                                         _Float16* __restrict__ d, int n) {
    int i = blockIdx.x * 256 + threadIdx.x;
    if (i < n) d[i] = (_Float16)s[i];
}

// processed_memory = enc @ mem_w.T   (M=B*T_ENC=16384, N=256, K=256), WMMA.
__global__ void __launch_bounds__(256) pm_gemm_kernel(const _Float16* __restrict__ enc16,
                                                      const _Float16* __restrict__ memw16,
                                                      float* __restrict__ pm) {
    int tid = threadIdx.x, wave = tid >> 5, lane = tid & 31;
    const _Float16* Arows = enc16 + (size_t)blockIdx.x * 16 * 256;
    for (int nt = wave; nt < 16; nt += 8) {
        v8f acc = {};
        for (int k0 = 0; k0 < 256; k0 += 32) {
            v16h a = load_frag(Arows, 256, k0, lane);
            v16h b = load_frag(memw16 + (size_t)(nt << 4) * 256, 256, k0, lane);
            acc = wmma32(a, b, acc);
        }
        int col = (nt << 4) + (lane & 15);
        int mb  = (lane >> 4) << 3;
#pragma unroll
        for (int r = 0; r < 8; ++r)
            pm[((size_t)blockIdx.x * 16 + mb + r) * 256 + col] = acc[r];
    }
}

// ---------------------------------------------------------------------------
// Persistent decoder: 4 blocks x 16 batch rows x 32 waves, 200 sequential steps.
// ---------------------------------------------------------------------------
struct DecArgs {
    const float* enc; const float* inputs; const int* memlen;
    const float* pre_b1; const float* pre_b2;
    const float* att_bih; const float* att_bhh;
    const float* v_w; const float* proj_b;
    const float* g1_bih; const float* g1_bhh;
    const float* g2_bih; const float* g2_bhh;
    const float* mel_b; const float* stop_w; const float* stop_b;
    const _Float16 *w_pre1, *w_pre2, *w_awih, *w_awhh, *w_q, *w_proj;
    const _Float16 *w_g1i, *w_g1h, *w_g2i, *w_g2h, *w_mel;
    const float* pm;
    float* out_mel; float* out_attn; float* out_stop;
};

__global__ void __launch_bounds__(NT) decoder_kernel(DecArgs A) {
    extern __shared__ char smem[];
    float* sh_hatt = (float*)smem;                 // 16x256 f32 state
    float* sh_ctx  = sh_hatt + 4096;
    float* sh_h1   = sh_ctx  + 4096;
    float* sh_h2   = sh_h1   + 4096;
    float* sh_d    = sh_h2   + 4096;               // query / proj / residual
    float* sh_al   = sh_d    + 4096;               // scores -> align
    _Float16* sh_a16 = (_Float16*)(sh_al + 4096);  // 16x512 A staging
    _Float16* sh_h16 = sh_a16 + 16 * 512;          // 16x256 h staging
    _Float16* sh_t16 = sh_h16 + 16 * 256;          // 16x256 prenet1 staging
    float* sh_xf32 = (float*)(sh_t16 + 16 * 256);  // 2 x 16x160 TDM frame buffers
    float* sh_v    = sh_xf32 + 2 * 16 * MELR;      // 256

    const int tid  = threadIdx.x;
    const int wave = tid >> 5, lane = tid & 31;
    const int m0   = blockIdx.x * 16;
    const int colT = lane & 15;
    const int mbT  = (lane >> 4) << 3;

    for (int i = tid; i < 4096; i += NT) {
        sh_hatt[i] = 0.f; sh_ctx[i] = 0.f; sh_h1[i] = 0.f; sh_h2[i] = 0.f;
    }
    if (tid < 256) sh_v[tid] = A.v_w[tid];
    __syncthreads();

#if HAVE_TDM
    const unsigned xbuf_base = (unsigned)(uintptr_t)(void*)sh_xf32;  // LDS byte offset
    if (wave == 0)                                                   // frame 0 -> buf 0
        tdm_load_frame(A.inputs + (size_t)m0 * T_DEC * MELR, xbuf_base);
#endif

    for (int t = 0; t < T_DEC; ++t) {
        // --- stage x (teacher forcing: prev frame, zeros at t=0) -----------
        if (t == 0) {
            for (int idx = tid; idx < 16 * MELR; idx += NT)
                sh_a16[(idx / MELR) * 512 + (idx % MELR)] = (_Float16)0.f;
        } else {
#if HAVE_TDM
            if (wave == 0) __builtin_amdgcn_s_wait_tensorcnt(0);
            __syncthreads();
            const float* xf = sh_xf32 + ((t - 1) & 1) * (16 * MELR);
            for (int idx = tid; idx < 16 * MELR; idx += NT) {
                int m = idx / MELR, c = idx - m * MELR;
                sh_a16[m * 512 + c] = (_Float16)xf[idx];
            }
#else
            for (int idx = tid; idx < 16 * MELR; idx += NT) {
                int m = idx / MELR, c = idx - m * MELR;
                sh_a16[m * 512 + c] =
                    (_Float16)A.inputs[((size_t)(m0 + m) * T_DEC + (t - 1)) * MELR + c];
            }
#endif
        }
        __syncthreads();
#if HAVE_TDM
        if (t + 1 < T_DEC && wave == 0)   // prefetch frame t (used at step t+1)
            tdm_load_frame(A.inputs + ((size_t)m0 * T_DEC + t) * MELR,
                           xbuf_base + (unsigned)((t & 1) * 16 * MELR * 4));
#endif
        // --- prenet layer 1: N=256 K=160, relu, f16 out --------------------
        for (int nt = wave; nt < 16; nt += NW) {
            v8f acc = tile_gemm(sh_a16, 512, A.w_pre1, MELR, nt << 4, lane);
            int col = (nt << 4) + colT;
#pragma unroll
            for (int r = 0; r < 8; ++r)
                sh_t16[(mbT + r) * 256 + col] =
                    (_Float16)fmaxf(acc[r] + A.pre_b1[col], 0.f);
        }
        __syncthreads();
        // --- prenet layer 2 (N=128 K=256) + ctx concat + h_att staging -----
        for (int nt = wave; nt < 8; nt += NW) {
            v8f acc = tile_gemm(sh_t16, 256, A.w_pre2, 256, nt << 4, lane);
            int col = (nt << 4) + colT;
#pragma unroll
            for (int r = 0; r < 8; ++r)
                sh_a16[(mbT + r) * 512 + col] =
                    (_Float16)fmaxf(acc[r] + A.pre_b2[col], 0.f);
        }
        for (int idx = tid; idx < 4096; idx += NT) {
            int m = idx >> 8, c = idx & 255;
            sh_a16[m * 512 + 128 + c] = (_Float16)sh_ctx[idx];
            sh_h16[idx]               = (_Float16)sh_hatt[idx];
        }
        __syncthreads();
        // --- attention GRUCell (Kx=384) ------------------------------------
        gru_block(sh_a16, 512, 384, sh_h16, A.w_awih, A.w_awhh,
                  A.att_bih, A.att_bhh, sh_hatt, nullptr, wave, lane);
        __syncthreads();
        for (int idx = tid; idx < 4096; idx += NT)
            sh_h16[idx] = (_Float16)sh_hatt[idx];
        __syncthreads();
        // --- query layer: q = h_att @ q_w.T  -> sh_d -----------------------
        for (int nt = wave; nt < 16; nt += NW) {
            v8f acc = tile_gemm(sh_h16, 256, A.w_q, 256, nt << 4, lane);
            int col = (nt << 4) + colT;
#pragma unroll
            for (int r = 0; r < 8; ++r) sh_d[(mbT + r) * 256 + col] = acc[r];
        }
        __syncthreads();
        // --- Bahdanau scores (per-row, VALU+trans, float4 loads) -----------
        for (int e = 0; e < 4; ++e) {
            int idx = tid + (e << 10);
            int m = idx >> 8, te = idx & 255;
            float s;
            if (te >= A.memlen[m0 + m]) s = -1e30f;
            else {
                const float4* pmr =
                    (const float4*)(A.pm + ((size_t)(m0 + m) * T_ENC + te) * ATT);
                const float4* qr = (const float4*)(sh_d + m * 256);
                const float4* vv = (const float4*)sh_v;
                s = 0.f;
                for (int d = 0; d < 64; ++d) {
                    float4 p = pmr[d], q = qr[d], v = vv[d];
                    s += v.x * tanhf(p.x + q.x) + v.y * tanhf(p.y + q.y)
                       + v.z * tanhf(p.z + q.z) + v.w * tanhf(p.w + q.w);
                }
            }
            sh_al[idx] = s;
        }
        __syncthreads();
        // --- masked softmax per row (1 row per wave, waves 0..15) ----------
        if (wave < 16) {
            int m = wave;
            float mx = -3.4e38f;
            for (int j = lane; j < 256; j += 32) mx = fmaxf(mx, sh_al[m * 256 + j]);
            for (int o = 16; o > 0; o >>= 1) mx = fmaxf(mx, __shfl_xor(mx, o, 32));
            float sum = 0.f;
            for (int j = lane; j < 256; j += 32) {
                float ev = expf(sh_al[m * 256 + j] - mx);
                sh_al[m * 256 + j] = ev; sum += ev;
            }
            for (int o = 16; o > 0; o >>= 1) sum += __shfl_xor(sum, o, 32);
            float inv = 1.f / sum;
            for (int j = lane; j < 256; j += 32) {
                float a = sh_al[m * 256 + j] * inv;
                sh_al[m * 256 + j] = a;
                A.out_attn[((size_t)(m0 + m) * T_DEC + t) * T_ENC + j] = a;
            }
        }
        __syncthreads();
        // --- context = align @ enc (per-row matvec, VALU) ------------------
        for (int e = 0; e < 4; ++e) {
            int idx = tid + (e << 10);
            int m = idx >> 8, dc = idx & 255;
            const float* al = sh_al + m * 256;
            const float* er = A.enc + (size_t)(m0 + m) * T_ENC * 256 + dc;
            float s = 0.f;
            for (int te = 0; te < 256; ++te) s += al[te] * er[(size_t)te * 256];
            sh_ctx[m * 256 + dc] = s;
        }
        __syncthreads();
        // --- proj input = concat(h_att, ctx) f16 ---------------------------
        for (int idx = tid; idx < 16 * 512; idx += NT) {
            int m = idx >> 9, c = idx & 511;
            float v = (c < 256) ? sh_hatt[m * 256 + c] : sh_ctx[m * 256 + c - 256];
            sh_a16[idx] = (_Float16)v;
        }
        __syncthreads();
        // --- project_to_decoder_in: N=256 K=512 ----------------------------
        for (int nt = wave; nt < 16; nt += NW) {
            v8f acc = tile_gemm(sh_a16, 512, A.w_proj, 512, nt << 4, lane);
            int col = (nt << 4) + colT;
#pragma unroll
            for (int r = 0; r < 8; ++r)
                sh_d[(mbT + r) * 256 + col] = acc[r] + A.proj_b[col];
        }
        __syncthreads();
        // --- decoder GRU 1 (residual) --------------------------------------
        for (int idx = tid; idx < 4096; idx += NT) {
            int m = idx >> 8, c = idx & 255;
            sh_a16[m * 512 + c] = (_Float16)sh_d[idx];
            sh_h16[idx]         = (_Float16)sh_h1[idx];
        }
        __syncthreads();
        gru_block(sh_a16, 512, 256, sh_h16, A.w_g1i, A.w_g1h,
                  A.g1_bih, A.g1_bhh, sh_h1, sh_d, wave, lane);
        __syncthreads();
        // --- decoder GRU 2 (residual) --------------------------------------
        for (int idx = tid; idx < 4096; idx += NT) {
            int m = idx >> 8, c = idx & 255;
            sh_a16[m * 512 + c] = (_Float16)sh_d[idx];
            sh_h16[idx]         = (_Float16)sh_h2[idx];
        }
        __syncthreads();
        gru_block(sh_a16, 512, 256, sh_h16, A.w_g2i, A.w_g2h,
                  A.g2_bih, A.g2_bhh, sh_h2, sh_d, wave, lane);
        __syncthreads();
        // --- p = concat(d, ctx) f16 ----------------------------------------
        for (int idx = tid; idx < 16 * 512; idx += NT) {
            int m = idx >> 9, c = idx & 511;
            float v = (c < 256) ? sh_d[m * 256 + c] : sh_ctx[m * 256 + c - 256];
            sh_a16[idx] = (_Float16)v;
        }
        __syncthreads();
        // --- mel head: N=160 K=512, direct global store --------------------
        for (int nt = wave; nt < 10; nt += NW) {
            v8f acc = tile_gemm(sh_a16, 512, A.w_mel, 512, nt << 4, lane);
            int col = (nt << 4) + colT;
#pragma unroll
            for (int r = 0; r < 8; ++r)
                A.out_mel[((size_t)(m0 + mbT + r) * T_DEC + t) * MELR + col] =
                    acc[r] + A.mel_b[col];
        }
        // --- stop token (1-wide, VALU) -------------------------------------
        if (tid < 16) {
            int m = tid;
            float s = A.stop_b[0];
            for (int k = 0; k < 256; ++k) s += sh_d[m * 256 + k] * A.stop_w[k];
            for (int k = 0; k < 256; ++k) s += sh_ctx[m * 256 + k] * A.stop_w[256 + k];
            s = 1.f / (1.f + expf(-s));
            A.out_stop[(size_t)(m0 + m) * (T_DEC * 2) + 2 * t]     = s;
            A.out_stop[(size_t)(m0 + m) * (T_DEC * 2) + 2 * t + 1] = s;
        }
        __syncthreads();
    }
}

// ---------------------------------------------------------------------------
extern "C" void kernel_launch(void* const* d_in, const int* in_sizes, int n_in,
                              void* d_out, int out_size, void* d_ws, size_t ws_size,
                              hipStream_t stream) {
    (void)in_sizes; (void)n_in; (void)out_size; (void)ws_size;
    const float* enc     = (const float*)d_in[0];
    const float* inputs  = (const float*)d_in[1];
    const int*   memlen  = (const int*)d_in[2];
    const float* pre_w1  = (const float*)d_in[3];
    const float* pre_b1  = (const float*)d_in[4];
    const float* pre_w2  = (const float*)d_in[5];
    const float* pre_b2  = (const float*)d_in[6];
    const float* mem_w   = (const float*)d_in[7];
    const float* att_wih = (const float*)d_in[8];
    const float* att_whh = (const float*)d_in[9];
    const float* att_bih = (const float*)d_in[10];
    const float* att_bhh = (const float*)d_in[11];
    const float* q_w     = (const float*)d_in[12];
    const float* v_w     = (const float*)d_in[13];
    const float* proj_w  = (const float*)d_in[14];
    const float* proj_b  = (const float*)d_in[15];
    const float* g1_wih  = (const float*)d_in[16];
    const float* g1_whh  = (const float*)d_in[17];
    const float* g1_bih  = (const float*)d_in[18];
    const float* g1_bhh  = (const float*)d_in[19];
    const float* g2_wih  = (const float*)d_in[20];
    const float* g2_whh  = (const float*)d_in[21];
    const float* g2_bih  = (const float*)d_in[22];
    const float* g2_bhh  = (const float*)d_in[23];
    const float* mel_w   = (const float*)d_in[24];
    const float* mel_b   = (const float*)d_in[25];
    const float* stop_w  = (const float*)d_in[26];
    const float* stop_b  = (const float*)d_in[27];

    char* ws = (char*)d_ws;
    size_t off = 0;
    auto take16 = [&](size_t n) {
        _Float16* p = (_Float16*)(ws + off);
        off += n * sizeof(_Float16);
        return p;
    };
    _Float16* enc16 = take16((size_t)BB * T_ENC * 256);
    float* pm = (float*)(ws + off);
    off += (size_t)BB * T_ENC * ATT * sizeof(float);
    _Float16* w_mem  = take16(256 * 256);
    _Float16* w_pre1 = take16(256 * 160);
    _Float16* w_pre2 = take16(128 * 256);
    _Float16* w_awih = take16(768 * 384);
    _Float16* w_awhh = take16(768 * 256);
    _Float16* w_q    = take16(256 * 256);
    _Float16* w_proj = take16(256 * 512);
    _Float16* w_g1i  = take16(768 * 256);
    _Float16* w_g1h  = take16(768 * 256);
    _Float16* w_g2i  = take16(768 * 256);
    _Float16* w_g2h  = take16(768 * 256);
    _Float16* w_mel  = take16(160 * 512);

    auto conv = [&](const float* s, _Float16* d, int n) {
        f32_to_f16_kernel<<<dim3((n + 255) / 256), dim3(256), 0, stream>>>(s, d, n);
    };
    conv(enc,     enc16,  BB * T_ENC * 256);
    conv(mem_w,   w_mem,  256 * 256);
    conv(pre_w1,  w_pre1, 256 * 160);
    conv(pre_w2,  w_pre2, 128 * 256);
    conv(att_wih, w_awih, 768 * 384);
    conv(att_whh, w_awhh, 768 * 256);
    conv(q_w,     w_q,    256 * 256);
    conv(proj_w,  w_proj, 256 * 512);
    conv(g1_wih,  w_g1i,  768 * 256);
    conv(g1_whh,  w_g1h,  768 * 256);
    conv(g2_wih,  w_g2i,  768 * 256);
    conv(g2_whh,  w_g2h,  768 * 256);
    conv(mel_w,   w_mel,  160 * 512);

    pm_gemm_kernel<<<dim3((BB * T_ENC) / 16), dim3(256), 0, stream>>>(enc16, w_mem, pm);

    float* out = (float*)d_out;
    DecArgs a;
    a.enc = enc; a.inputs = inputs; a.memlen = memlen;
    a.pre_b1 = pre_b1; a.pre_b2 = pre_b2;
    a.att_bih = att_bih; a.att_bhh = att_bhh;
    a.v_w = v_w; a.proj_b = proj_b;
    a.g1_bih = g1_bih; a.g1_bhh = g1_bhh;
    a.g2_bih = g2_bih; a.g2_bhh = g2_bhh;
    a.mel_b = mel_b; a.stop_w = stop_w; a.stop_b = stop_b;
    a.w_pre1 = w_pre1; a.w_pre2 = w_pre2;
    a.w_awih = w_awih; a.w_awhh = w_awhh;
    a.w_q = w_q; a.w_proj = w_proj;
    a.w_g1i = w_g1i; a.w_g1h = w_g1h; a.w_g2i = w_g2i; a.w_g2h = w_g2h;
    a.w_mel = w_mel;
    a.pm = pm;
    a.out_mel  = out;
    a.out_attn = out + (size_t)BB * T_DEC * MELR;
    a.out_stop = a.out_attn + (size_t)BB * T_DEC * T_ENC;

    size_t shmem = 6 * 4096 * sizeof(float)
                 + (16 * 512 + 16 * 256 + 16 * 256) * sizeof(_Float16)
                 + 2 * 16 * MELR * sizeof(float)
                 + 256 * sizeof(float);
    decoder_kernel<<<dim3(BB / 16), dim3(NT), shmem, stream>>>(a);
}
